// Attention_v1_29807073034392
// MI455X (gfx1250) — matrix operand order repaired and checked by reading, not compiled, not run on hardware
//
#include <hip/hip_runtime.h>

typedef __attribute__((ext_vector_type(16))) _Float16 v16h;
typedef __attribute__((ext_vector_type(8)))  float    v8f;
typedef __attribute__((ext_vector_type(8)))  _Float16 v8h;

#define EMBD  1024
#define NSEQ  2048
#define BATCH 2
#define NHEAD 8        // value heads
#define NSUB  16       // 2*H q/k sub-heads
#define HDIM  64       // sub-head dim
#define VDIM  128      // value head dim
#define MROWS (BATCH*NSEQ)   // 4096 rows in projection GEMMs

// ---------------------------------------------------------------------------
// bulk fp32 -> f16 conversion (8 elements/thread, 2x b128 load + b128 store)
// ---------------------------------------------------------------------------
__global__ __launch_bounds__(256)
void cvt_kernel(const float* __restrict__ in, _Float16* __restrict__ out, int n8)
{
  const int i = blockIdx.x * blockDim.x + threadIdx.x;
  if (i < n8) {
    v8f x = ((const v8f*)in)[i];
    ((v8h*)out)[i] = __builtin_convertvector(x, v8h);
  }
}

// ---------------------------------------------------------------------------
// lambda = exp(sum lq1*lk1) - exp(sum lq2*lk2) + 0.8
// ---------------------------------------------------------------------------
__global__ void lambda_kernel(const float* __restrict__ lq1, const float* __restrict__ lk1,
                              const float* __restrict__ lq2, const float* __restrict__ lk2,
                              float* __restrict__ lam)
{
  __shared__ float s1[64], s2[64];
  int t = threadIdx.x;
  s1[t] = lq1[t] * lk1[t];
  s2[t] = lq2[t] * lk2[t];
  __syncthreads();
  for (int off = 32; off > 0; off >>= 1) {
    if (t < off) { s1[t] += s1[t + off]; s2[t] += s2[t + off]; }
    __syncthreads();
  }
  if (t == 0) lam[0] = __expf(s1[0]) - __expf(s2[0]) + 0.8f;
}

// ---------------------------------------------------------------------------
// NT GEMM (all-f16 operands): C[m,o] = sum_k A[m,k]*W[o,k] + bias[o]
// Each wave computes a 16x64 tile via 4 f32 accumulators, K-loop step 32.
// Hot loop = pure global_load_b128 + v_wmma.
// MODE 0: f16 store to Qh/Kh layout [B, 2H, N, 64]
// MODE 1: f16 store to Vt layout    [B, H, 128, N]  (transposed for P*V frags)
// MODE 2: f32 store to d_out        [M, 1024]
// ---------------------------------------------------------------------------
template<int MODE>
__global__ __launch_bounds__(256)
void proj_gemm(const _Float16* __restrict__ A, const _Float16* __restrict__ W,
               const float* __restrict__ bias, void* __restrict__ outv)
{
  const int lane = threadIdx.x & 31;
  const int wave = threadIdx.x >> 5;
  const int lo = lane & 15;
  const int hi = lane >> 4;
  const int m0 = blockIdx.x * 128 + wave * 16;
  const int n0 = blockIdx.y * 64;

  v8f acc[4];
#pragma unroll
  for (int f = 0; f < 4; ++f)
#pragma unroll
    for (int r = 0; r < 8; ++r) acc[f][r] = 0.0f;

  const _Float16* arow = A + (size_t)(m0 + lo) * EMBD + hi * 8;

  for (int k0 = 0; k0 < EMBD; k0 += 32) {
    // A fragment (16x32): lane lo = row, halves 0..7 -> K=hi*8+0..7, 8..15 -> K=16+hi*8+0..7
    v16h a;
    const _Float16* ar = arow + k0;
#pragma unroll
    for (int i = 0; i < 8; ++i) { a[i] = ar[i]; a[i + 8] = ar[16 + i]; }
    // 4 B fragments (32x16): lane lo = out-channel column, halves 0..7 -> K=hi*8+0..7, 8..15 -> K=16+hi*8+0..7
#pragma unroll
    for (int f = 0; f < 4; ++f) {
      const _Float16* wr = W + (size_t)(n0 + f * 16 + lo) * EMBD + k0 + hi * 8;
      v16h b;
#pragma unroll
      for (int i = 0; i < 8; ++i) { b[i] = wr[i]; b[i + 8] = wr[16 + i]; }
      acc[f] = __builtin_amdgcn_wmma_f32_16x16x32_f16(false, a, false, b,
                                                      (short)0, acc[f], false, false);
    }
  }

  // bias + store (C layout: VGPR r holds row m0 + r + 8*hi, column = tile col lo)
#pragma unroll
  for (int f = 0; f < 4; ++f) {
    const int ocol = n0 + f * 16 + lo;
    const float bv = bias[ocol];
#pragma unroll
    for (int r = 0; r < 8; ++r) {
      const float val = acc[f][r] + bv;
      const int mrow = m0 + r + hi * 8;
      const int bb = mrow >> 11;        // /2048
      const int nn = mrow & 2047;
      if (MODE == 0) {                  // [B,2H,N,64]
        const int sh = ocol >> 6, d = ocol & 63;
        ((_Float16*)outv)[(((bb * NSUB) + sh) * NSEQ + nn) * HDIM + d] = (_Float16)val;
      } else if (MODE == 1) {           // [B,H,128,N] transposed
        const int h = ocol >> 7, d = ocol & 127;
        ((_Float16*)outv)[(((bb * NHEAD) + h) * VDIM + d) * NSEQ + nn] = (_Float16)val;
      } else {                          // f32 [M,1024]
        ((float*)outv)[(size_t)mrow * EMBD + ocol] = val;
      }
    }
  }
}

// ---------------------------------------------------------------------------
// Flash attention per sub-head: each wave owns one 16-row Q tile, streams all
// 2048 kv positions in chunks of 32 with online softmax.
// Osub layout: [B, 2H, N, 128] fp32 (already normalized by l).
// ---------------------------------------------------------------------------
__global__ __launch_bounds__(256)
void attn_kernel(const _Float16* __restrict__ Qh, const _Float16* __restrict__ Kh,
                 const _Float16* __restrict__ Vt, float* __restrict__ Osub)
{
  __shared__ _Float16 Psh[8][16 * 32];   // per-wave P transpose staging (1KB each)

  const int lane = threadIdx.x & 31;
  const int wave = threadIdx.x >> 5;
  const int lo = lane & 15;
  const int hi = lane >> 4;

  const int bsh = blockIdx.x;            // 0..31 : (b, sub-head)
  const int b   = bsh >> 4;
  const int sh  = bsh & 15;
  const int h   = sh >> 1;               // value head
  const int m0  = blockIdx.y * 128 + wave * 16;

  const _Float16* Qb = Qh + (size_t)bsh * NSEQ * HDIM;
  const _Float16* Kb = Kh + (size_t)bsh * NSEQ * HDIM;
  const _Float16* Vb = Vt + (size_t)(b * NHEAD + h) * VDIM * NSEQ;

  // Q fragments for K-dim 0..31 and 32..63
  v16h qa[2];
#pragma unroll
  for (int kb = 0; kb < 2; ++kb) {
    const _Float16* qr = Qb + (size_t)(m0 + lo) * HDIM + kb * 32 + hi * 8;
#pragma unroll
    for (int i = 0; i < 8; ++i) { qa[kb][i] = qr[i]; qa[kb][i + 8] = qr[16 + i]; }
  }

  v8f o[8];
  float mrow[8], lrow[8];
#pragma unroll
  for (int f = 0; f < 8; ++f)
#pragma unroll
    for (int r = 0; r < 8; ++r) o[f][r] = 0.0f;
#pragma unroll
  for (int r = 0; r < 8; ++r) { mrow[r] = -1e30f; lrow[r] = 0.0f; }

  const float scale = 0.125f;            // 1/sqrt(64)

  for (int j = 0; j < NSEQ; j += 32) {
    // ---- S = Q * K^T for 32 kv columns (two 16-col tiles) ----
    v8f s01[2];
#pragma unroll
    for (int t = 0; t < 2; ++t) {
      v8f s;
#pragma unroll
      for (int r = 0; r < 8; ++r) s[r] = 0.0f;
#pragma unroll
      for (int kb = 0; kb < 2; ++kb) {
        const _Float16* kr = Kb + (size_t)(j + t * 16 + lo) * HDIM + kb * 32 + hi * 8;
        v16h kf;
#pragma unroll
        for (int i = 0; i < 8; ++i) { kf[i] = kr[i]; kf[i + 8] = kr[16 + i]; }
        s = __builtin_amdgcn_wmma_f32_16x16x32_f16(false, qa[kb], false, kf,
                                                   (short)0, s, false, false);
      }
      s01[t] = s;
    }
#pragma unroll
    for (int r = 0; r < 8; ++r) { s01[0][r] *= scale; s01[1][r] *= scale; }

    // ---- online softmax: row max over 16 lanes of each half ----
    float rmax[8];
#pragma unroll
    for (int r = 0; r < 8; ++r) rmax[r] = fmaxf(s01[0][r], s01[1][r]);
#pragma unroll
    for (int off = 1; off < 16; off <<= 1)
#pragma unroll
      for (int r = 0; r < 8; ++r) rmax[r] = fmaxf(rmax[r], __shfl_xor(rmax[r], off, 32));

    float alpha[8];
#pragma unroll
    for (int r = 0; r < 8; ++r) {
      const float mn = fmaxf(mrow[r], rmax[r]);
      alpha[r] = __expf(mrow[r] - mn);
      mrow[r] = mn;
    }
#pragma unroll
    for (int r = 0; r < 8; ++r) {
      s01[0][r] = __expf(s01[0][r] - mrow[r]);
      s01[1][r] = __expf(s01[1][r] - mrow[r]);
    }
    float rsum[8];
#pragma unroll
    for (int r = 0; r < 8; ++r) rsum[r] = s01[0][r] + s01[1][r];
#pragma unroll
    for (int off = 1; off < 16; off <<= 1)
#pragma unroll
      for (int r = 0; r < 8; ++r) rsum[r] += __shfl_xor(rsum[r], off, 32);
#pragma unroll
    for (int r = 0; r < 8; ++r) lrow[r] = lrow[r] * alpha[r] + rsum[r];

    // rescale O accumulators
#pragma unroll
    for (int f = 0; f < 8; ++f)
#pragma unroll
      for (int r = 0; r < 8; ++r) o[f][r] *= alpha[r];

    // ---- P (C layout) -> LDS -> A fragment (16x32) ----
    _Float16* P = Psh[wave];
#pragma unroll
    for (int r = 0; r < 8; ++r) {
      const int row = r + hi * 8;
      P[row * 32 + lo]      = (_Float16)s01[0][r];
      P[row * 32 + 16 + lo] = (_Float16)s01[1][r];
    }
    v16h pa;
    {
      const _Float16* pr = P + lo * 32 + hi * 8;
#pragma unroll
      for (int i = 0; i < 8; ++i) { pa[i] = pr[i]; pa[i + 8] = pr[16 + i]; }
    }

    // ---- O += P * V  (8 column tiles of the 128-wide value head) ----
#pragma unroll
    for (int f = 0; f < 8; ++f) {
      const _Float16* vr = Vb + (size_t)(f * 16 + lo) * NSEQ + j + hi * 8;
      v16h vf;
#pragma unroll
      for (int i = 0; i < 8; ++i) { vf[i] = vr[i]; vf[i + 8] = vr[16 + i]; }
      o[f] = __builtin_amdgcn_wmma_f32_16x16x32_f16(false, pa, false, vf,
                                                    (short)0, o[f], false, false);
    }
  }

  // ---- normalize and store ----
#pragma unroll
  for (int f = 0; f < 8; ++f) {
#pragma unroll
    for (int r = 0; r < 8; ++r) {
      const int row = m0 + r + hi * 8;
      const int col = f * 16 + lo;
      Osub[((size_t)bsh * NSEQ + row) * VDIM + col] = o[f][r] / lrow[r];
    }
  }
}

// ---------------------------------------------------------------------------
// diff combine: Of16[b,n,h*128+d] = Osub[b,2h,n,d] - lam*Osub[b,2h+1,n,d]
// ---------------------------------------------------------------------------
__global__ __launch_bounds__(256)
void combine_kernel(const float* __restrict__ Osub, const float* __restrict__ lamp,
                    _Float16* __restrict__ Of16)
{
  const int idx = blockIdx.x * blockDim.x + threadIdx.x;   // B*H*N*128 = 4194304
  const float lam = lamp[0];
  const int d = idx & 127;
  const int n = (idx >> 7) & 2047;
  const int h = (idx >> 18) & 7;
  const int b = idx >> 21;
  const float v0 = Osub[(((size_t)(b * NSUB + 2 * h)     * NSEQ) + n) * VDIM + d];
  const float v1 = Osub[(((size_t)(b * NSUB + 2 * h + 1) * NSEQ) + n) * VDIM + d];
  Of16[((size_t)(b * NSEQ + n)) * EMBD + h * VDIM + d] = (_Float16)(v0 - lam * v1);
}

// ---------------------------------------------------------------------------
extern "C" void kernel_launch(void* const* d_in, const int* in_sizes, int n_in,
                              void* d_out, int out_size, void* d_ws, size_t ws_size,
                              hipStream_t stream)
{
  (void)in_sizes; (void)n_in; (void)out_size; (void)ws_size;
  const float* q   = (const float*)d_in[0];
  const float* k   = (const float*)d_in[1];
  const float* v   = (const float*)d_in[2];
  const float* Wq  = (const float*)d_in[3];
  const float* bq  = (const float*)d_in[4];
  const float* Wk  = (const float*)d_in[5];
  const float* bk  = (const float*)d_in[6];
  const float* Wv  = (const float*)d_in[7];
  const float* bv  = (const float*)d_in[8];
  const float* Wo  = (const float*)d_in[9];
  const float* bo  = (const float*)d_in[10];
  const float* lq1 = (const float*)d_in[11];
  const float* lk1 = (const float*)d_in[12];
  const float* lq2 = (const float*)d_in[13];
  const float* lk2 = (const float*)d_in[14];

  const size_t MB = 1048576;
  char* ws = (char*)d_ws;
  _Float16* Qh   = (_Float16*)(ws);              //  8 MB  [B,2H,N,64]
  _Float16* Kh   = (_Float16*)(ws +  8 * MB);    //  8 MB
  _Float16* Vt   = (_Float16*)(ws + 16 * MB);    //  8 MB  [B,H,128,N]
  float*    Osub = (float*)   (ws + 24 * MB);    // 32 MB  [B,2H,N,128]
  _Float16* Xq   = (_Float16*)(ws + 56 * MB);    //  8 MB  (dead after Q proj)
  _Float16* Of16 = (_Float16*)(ws + 56 * MB);    //  8 MB  aliases Xq
  _Float16* Xk   = (_Float16*)(ws + 64 * MB);    //  8 MB
  _Float16* Xv   = (_Float16*)(ws + 72 * MB);    //  8 MB
  _Float16* Wqh  = (_Float16*)(ws + 80 * MB);    //  2 MB
  _Float16* Wkh  = (_Float16*)(ws + 82 * MB);
  _Float16* Wvh  = (_Float16*)(ws + 84 * MB);
  _Float16* Woh  = (_Float16*)(ws + 86 * MB);
  float*    lam  = (float*)   (ws + 88 * MB);

  lambda_kernel<<<1, 64, 0, stream>>>(lq1, lk1, lq2, lk2, lam);

  // fp32 -> f16 pre-conversion (one pass; hot GEMM loops become pure load+wmma)
  const int nX8 = (MROWS * EMBD) / 8;            // 524288
  const int nW8 = (EMBD * EMBD) / 8;             // 131072
  cvt_kernel<<<nX8 / 256, 256, 0, stream>>>(q, Xq, nX8);
  cvt_kernel<<<nX8 / 256, 256, 0, stream>>>(k, Xk, nX8);
  cvt_kernel<<<nX8 / 256, 256, 0, stream>>>(v, Xv, nX8);
  cvt_kernel<<<nW8 / 256, 256, 0, stream>>>(Wq, Wqh, nW8);
  cvt_kernel<<<nW8 / 256, 256, 0, stream>>>(Wk, Wkh, nW8);
  cvt_kernel<<<nW8 / 256, 256, 0, stream>>>(Wv, Wvh, nW8);
  cvt_kernel<<<nW8 / 256, 256, 0, stream>>>(Wo, Woh, nW8);

  const dim3 ggrid(MROWS / 128, EMBD / 64);
  proj_gemm<0><<<ggrid, 256, 0, stream>>>(Xq, Wqh, bq, Qh);
  proj_gemm<0><<<ggrid, 256, 0, stream>>>(Xk, Wkh, bk, Kh);
  proj_gemm<1><<<ggrid, 256, 0, stream>>>(Xv, Wvh, bv, Vt);

  attn_kernel<<<dim3(BATCH * NSUB, NSEQ / 128), 256, 0, stream>>>(Qh, Kh, Vt, Osub);

  combine_kernel<<<(BATCH * NHEAD * NSEQ * VDIM) / 256, 256, 0, stream>>>(Osub, lam, Of16);

  proj_gemm<2><<<ggrid, 256, 0, stream>>>(Of16, Woh, bo, (float*)d_out);
}
